// GraphDecoderLayer_76373108457772
// MI455X (gfx1250) — matrix operand order repaired and checked by reading, not compiled, not run on hardware
//
#include <hip/hip_runtime.h>

typedef _Float16 h16;
typedef __attribute__((ext_vector_type(16))) _Float16 v16h;
typedef __attribute__((ext_vector_type(8)))  _Float16 v8h;
typedef __attribute__((ext_vector_type(8)))  float    v8f;
typedef __attribute__((ext_vector_type(4)))  float    v4f;

#define DD 128   // feature dim

// ---- helpers -------------------------------------------------------------

static __device__ __forceinline__ v8h cvt8(const float* p) {
  v4f a = *(const v4f*)p;
  v4f b = *(const v4f*)(p + 4);
  v8h r;
  r[0]=(h16)a[0]; r[1]=(h16)a[1]; r[2]=(h16)a[2]; r[3]=(h16)a[3];
  r[4]=(h16)b[0]; r[5]=(h16)b[1]; r[6]=(h16)b[2]; r[7]=(h16)b[3];
  return r;
}

// B fragment from fragment-swizzled f16 weights:
// wswz[((kk*128 + n)*2 + half)*16 + i] == W[kk*32 + (i>>3)*16 + half*8 + (i&7)][n]
// -> per-lane contiguous 32B load; 32 lanes cover a contiguous 1KB block.
static __device__ __forceinline__ v16h load_b_frag(const h16* w, int kk, int n, int half) {
  const h16* p = w + ((((size_t)kk * DD + n) * 2 + half) << 4);
  return *(const v16h*)p;
}

static __device__ __forceinline__ v8f wmma16(v16h a, v16h b, v8f c) {
  return __builtin_amdgcn_wmma_f32_16x16x32_f16(false, a, false, b, (short)0, c,
                                                false, false);
}

// A fragment (16x32 f16) for K-step kk from two f32 row pointers
// (lorow = concat K 0..127, hirow = concat K 128..255).
static __device__ __forceinline__ v16h load_a_2src(const float* lorow,
                                                   const float* hirow,
                                                   int kk, int kbA) {
  const int s1 = kk * 32 + kbA;        // 8-aligned, never crosses the 128 split
  const int s2 = s1 + 16;
  const float* p1 = (s1 < DD) ? lorow + s1 : hirow + (s1 - DD);
  const float* p2 = (s2 < DD) ? lorow + s2 : hirow + (s2 - DD);
  v8h lo = cvt8(p1), hi = cvt8(p2);
  v16h a;
#pragma unroll
  for (int i = 0; i < 8; ++i) { a[i] = lo[i]; a[i + 8] = hi[i]; }
  return a;
}

// A fragment for GEMM2, read back from the per-wave LDS staging row.
static __device__ __forceinline__ v16h load_a_lds(const h16* arow, int kk, int kbA) {
  const h16* p = arow + kk * 32 + kbA;
  v8h lo = *(const v8h*)p;
  v8h hi = *(const v8h*)(p + 16);
  v16h a;
#pragma unroll
  for (int i = 0; i < 8; ++i) { a[i] = lo[i]; a[i + 8] = hi[i]; }
  return a;
}

static __device__ __forceinline__ void relu_store(h16* st, v8f c, float bias,
                                                  int col, int half) {
#pragma unroll
  for (int r = 0; r < 8; ++r) {
    float v = c[r] + bias;
    v = v > 0.f ? v : 0.f;
    st[(r + half * 8) * DD + col] = (h16)v;
  }
}

// ---- prep: f32 weights -> fragment-swizzled f16 in workspace -------------
// ws halves: [0,32768) w_m1 | [32768,49152) w_m2 | [49152,81920) w_u1 | [81920,98304) w_u2
__global__ void prep_weights(const float* __restrict__ wm1, const float* __restrict__ wm2,
                             const float* __restrict__ wu1, const float* __restrict__ wu2,
                             h16* __restrict__ dst) {
  int tid = blockIdx.x * blockDim.x + threadIdx.x;
  if (tid >= 98304) return;
  const float* src; int idx;
  if (tid < 32768)      { src = wm1; idx = tid; }
  else if (tid < 49152) { src = wm2; idx = tid - 32768; }
  else if (tid < 81920) { src = wu1; idx = tid - 49152; }
  else                  { src = wu2; idx = tid - 81920; }
  const int kk  = idx >> 12;        // / 4096
  const int rem = idx & 4095;
  const int n   = rem >> 5;
  const int r   = rem & 31;         // = half*16 + i
  const int K   = kk * 32 + ((r & 8) << 1) + ((r >> 4) << 3) + (r & 7);
  dst[tid] = (h16)src[K * DD + n];
}

__global__ void zero_f32(float* p, long n) {
  long i = (long)blockIdx.x * blockDim.x + threadIdx.x;
  const long stride = (long)gridDim.x * blockDim.x;
  for (; i < n; i += stride) p[i] = 0.f;
}

// ---- kernel 1: per-edge message MLP + atomic scatter-add -----------------
__global__ void __launch_bounds__(256) msg_mlp_scatter(
    const float* __restrict__ x, const int* __restrict__ erow,
    const int* __restrict__ ecol,
    const h16* __restrict__ w1s, const h16* __restrict__ w2s,
    const float* __restrict__ b1, const float* __restrict__ b2,
    float* agg, int nE) {
  __shared__ __align__(16) h16 stage[8 * 16 * DD];   // 4 KB per wave
  const int lane = threadIdx.x & 31;
  const int wave = threadIdx.x >> 5;
  const int half = lane >> 4;
  const int m    = lane & 15;
  const int kbA  = half * 8;
  h16* st = stage + wave * (16 * DD);
  const int gw = blockIdx.x * 8 + wave;
  const int nw = gridDim.x * 8;
  const int ntiles = (nE + 15) >> 4;
  const v8f vz = {0.f, 0.f, 0.f, 0.f, 0.f, 0.f, 0.f, 0.f};

  for (int tile = gw; tile < ntiles; tile += nw) {
    const int e0 = tile << 4;
    int em = e0 + m; if (em >= nE) em = nE - 1;
    const float* lorow = x + (size_t)erow[em] * DD;   // concat K 0..127
    const float* hirow = x + (size_t)ecol[em] * DD;   // concat K 128..255

    // GEMM1: [16,256]@[256,128] in two 64-column passes (keeps VGPRs low)
#pragma unroll
    for (int pass = 0; pass < 2; ++pass) {
      const int nb = pass * 64 + m;
      v8f c0 = vz, c1 = vz, c2 = vz, c3 = vz;
#pragma unroll 1
      for (int kk = 0; kk < 8; ++kk) {
        v16h a = load_a_2src(lorow, hirow, kk, kbA);
        c0 = wmma16(a, load_b_frag(w1s, kk, nb,      half), c0);
        c1 = wmma16(a, load_b_frag(w1s, kk, nb + 16, half), c1);
        c2 = wmma16(a, load_b_frag(w1s, kk, nb + 32, half), c2);
        c3 = wmma16(a, load_b_frag(w1s, kk, nb + 48, half), c3);
      }
      relu_store(st, c0, b1[nb],      nb,      half);
      relu_store(st, c1, b1[nb + 16], nb + 16, half);
      relu_store(st, c2, b1[nb + 32], nb + 32, half);
      relu_store(st, c3, b1[nb + 48], nb + 48, half);
    }
    asm volatile("s_wait_dscnt 0x0" ::: "memory");   // same-wave LDS RAW fence

    // destination node ids for this edge tile
    int dsts[8];
#pragma unroll
    for (int r = 0; r < 8; ++r) {
      const int e = e0 + r + half * 8;
      dsts[r] = (e < nE) ? ecol[e] : -1;
    }

    // GEMM2: [16,128]@[128,128] in two 64-column passes, scatter per pass
    const h16* arow = st + m * DD;
#pragma unroll
    for (int pass = 0; pass < 2; ++pass) {
      const int nb = pass * 64 + m;
      v8f c0 = vz, c1 = vz, c2 = vz, c3 = vz;
#pragma unroll 1
      for (int kk = 0; kk < 4; ++kk) {
        v16h a = load_a_lds(arow, kk, kbA);
        c0 = wmma16(a, load_b_frag(w2s, kk, nb,      half), c0);
        c1 = wmma16(a, load_b_frag(w2s, kk, nb + 16, half), c1);
        c2 = wmma16(a, load_b_frag(w2s, kk, nb + 32, half), c2);
        c3 = wmma16(a, load_b_frag(w2s, kk, nb + 48, half), c3);
      }
#pragma unroll
      for (int t = 0; t < 4; ++t) {
        const int n = nb + t * 16;
        const float bias = b2[n];
        const v8f c = (t == 0) ? c0 : (t == 1) ? c1 : (t == 2) ? c2 : c3;
#pragma unroll
        for (int r = 0; r < 8; ++r) {
          if (dsts[r] >= 0)
            atomicAdd(agg + (size_t)dsts[r] * DD + n, c[r] + bias);
        }
      }
    }
  }
}

// ---- kernel 2: node update MLP + LayerNorm + residual --------------------
// agg aliases out: each tile's agg rows are read before that same tile's
// out rows are written, and tiles partition the rows.
__global__ void __launch_bounds__(256) node_mlp_ln(
    const float* __restrict__ x, const float* agg,
    const h16* __restrict__ w1s, const h16* __restrict__ w2s,
    const float* __restrict__ b1, const float* __restrict__ b2,
    const float* __restrict__ gma, const float* __restrict__ bta,
    float* out, int nN) {
  __shared__ __align__(16) h16 stage[8 * 16 * DD];
  const int lane = threadIdx.x & 31;
  const int wave = threadIdx.x >> 5;
  const int half = lane >> 4;
  const int m    = lane & 15;
  const int kbA  = half * 8;
  h16* st = stage + wave * (16 * DD);
  const int gw = blockIdx.x * 8 + wave;
  const int nw = gridDim.x * 8;
  const int ntiles = (nN + 15) >> 4;
  const v8f vz = {0.f, 0.f, 0.f, 0.f, 0.f, 0.f, 0.f, 0.f};

  for (int tile = gw; tile < ntiles; tile += nw) {
    const int n0 = tile << 4;
    int nm = n0 + m; if (nm >= nN) nm = nN - 1;
    const float* lorow = x   + (size_t)nm * DD;   // concat K 0..127
    const float* hirow = agg + (size_t)nm * DD;   // concat K 128..255

#pragma unroll
    for (int pass = 0; pass < 2; ++pass) {
      const int nb = pass * 64 + m;
      v8f c0 = vz, c1 = vz, c2 = vz, c3 = vz;
#pragma unroll 1
      for (int kk = 0; kk < 8; ++kk) {
        v16h a = load_a_2src(lorow, hirow, kk, kbA);
        c0 = wmma16(a, load_b_frag(w1s, kk, nb,      half), c0);
        c1 = wmma16(a, load_b_frag(w1s, kk, nb + 16, half), c1);
        c2 = wmma16(a, load_b_frag(w1s, kk, nb + 32, half), c2);
        c3 = wmma16(a, load_b_frag(w1s, kk, nb + 48, half), c3);
      }
      relu_store(st, c0, b1[nb],      nb,      half);
      relu_store(st, c1, b1[nb + 16], nb + 16, half);
      relu_store(st, c2, b1[nb + 32], nb + 32, half);
      relu_store(st, c3, b1[nb + 48], nb + 48, half);
    }
    asm volatile("s_wait_dscnt 0x0" ::: "memory");

    // GEMM2: keep all 8 finished tiles for LayerNorm
    v8f hh[8];
    const h16* arow = st + m * DD;
#pragma unroll
    for (int pass = 0; pass < 2; ++pass) {
      const int nb = pass * 64 + m;
      v8f c0 = vz, c1 = vz, c2 = vz, c3 = vz;
#pragma unroll 1
      for (int kk = 0; kk < 4; ++kk) {
        v16h a = load_a_lds(arow, kk, kbA);
        c0 = wmma16(a, load_b_frag(w2s, kk, nb,      half), c0);
        c1 = wmma16(a, load_b_frag(w2s, kk, nb + 16, half), c1);
        c2 = wmma16(a, load_b_frag(w2s, kk, nb + 32, half), c2);
        c3 = wmma16(a, load_b_frag(w2s, kk, nb + 48, half), c3);
      }
#pragma unroll
      for (int r = 0; r < 8; ++r) {
        hh[pass * 4 + 0][r] = c0[r] + b2[nb];
        hh[pass * 4 + 1][r] = c1[r] + b2[nb + 16];
        hh[pass * 4 + 2][r] = c2[r] + b2[nb + 32];
        hh[pass * 4 + 3][r] = c3[r] + b2[nb + 48];
      }
    }

    // LayerNorm stats: row M = r + half*8 lives in element r of the 16 lanes
    // of this half-wave across the 8 N-tiles -> 4 shfl_xor steps per stat.
    float mu[8], rsg[8];
#pragma unroll
    for (int r = 0; r < 8; ++r) {
      float s = 0.f, q = 0.f;
#pragma unroll
      for (int t = 0; t < 8; ++t) { const float v = hh[t][r]; s += v; q += v * v; }
#pragma unroll
      for (int msk = 1; msk < 16; msk <<= 1) {
        s += __shfl_xor(s, msk, 32);
        q += __shfl_xor(q, msk, 32);
      }
      const float mean = s * (1.f / DD);
      const float var  = q * (1.f / DD) - mean * mean;
      mu[r]  = mean;
      rsg[r] = rsqrtf(var + 1e-5f);
    }

    // normalize * gamma + beta + residual, store
#pragma unroll
    for (int t = 0; t < 8; ++t) {
      const int n = t * 16 + m;
      const float g = gma[n], be = bta[n];
#pragma unroll
      for (int r = 0; r < 8; ++r) {
        const int node = n0 + r + half * 8;
        if (node < nN) {
          const float res = x[(size_t)node * DD + n];
          out[(size_t)node * DD + n] = (hh[t][r] - mu[r]) * rsg[r] * g + be + res;
        }
      }
    }
  }
}

// ---- launch --------------------------------------------------------------
extern "C" void kernel_launch(void* const* d_in, const int* in_sizes, int n_in,
                              void* d_out, int out_size, void* d_ws, size_t ws_size,
                              hipStream_t stream) {
  (void)n_in; (void)out_size; (void)ws_size;
  const float* x    = (const float*)d_in[0];
  const int*   ei   = (const int*)  d_in[1];
  const float* w_m1 = (const float*)d_in[2];
  const float* b_m1 = (const float*)d_in[3];
  const float* w_m2 = (const float*)d_in[4];
  const float* b_m2 = (const float*)d_in[5];
  const float* w_u1 = (const float*)d_in[6];
  const float* b_u1 = (const float*)d_in[7];
  const float* w_u2 = (const float*)d_in[8];
  const float* b_u2 = (const float*)d_in[9];
  const float* gma  = (const float*)d_in[10];
  const float* bta  = (const float*)d_in[11];
  const int nN = in_sizes[0] / DD;
  const int nE = in_sizes[1] / 2;
  float* out = (float*)d_out;       // doubles as agg scratch (zeroed each call)
  h16* ws = (h16*)d_ws;             // 196608 B of swizzled f16 weights

  prep_weights<<<(98304 + 255) / 256, 256, 0, stream>>>(w_m1, w_m2, w_u1, w_u2, ws);
  zero_f32<<<2048, 256, 0, stream>>>(out, (long)nN * DD);

  const int etiles = (nE + 15) >> 4;
  msg_mlp_scatter<<<(etiles + 7) / 8, 256, 0, stream>>>(
      x, ei, ei + nE, ws, ws + 32768, b_m1, b_m2, out, nE);

  const int ntile2 = (nN + 15) >> 4;
  node_mlp_ln<<<(ntile2 + 7) / 8, 256, 0, stream>>>(
      x, out, ws + 49152, ws + 81920, b_u1, b_u2, gma, bta, out, nN);
}
